// SingleAttention_54924041781712
// MI455X (gfx1250) — matrix operand order repaired and checked by reading, not compiled, not run on hardware
//
#include <hip/hip_runtime.h>
#include <math.h>

// ---------------------------------------------------------------------------
// SingleAttention for MI455X (gfx1250), compile-only tuning.
//   Round 3: double-buffered TDM staging — tile i+1's tensor_load_to_lds is
//   issued before computing tile i; s_wait_tensorcnt(2) retires only the
//   oldest (in-order) pair, overlapping DMA with WMMA+softmax compute.
// Scratch (d_ws): Qh[8192*64] f16, Kh[8192*64] f16, Vt[4*64*2048] f16 = 3 MB.
// ---------------------------------------------------------------------------

typedef __attribute__((ext_vector_type(16))) _Float16 v16h;
typedef __attribute__((ext_vector_type(8)))  _Float16 v8h;
typedef __attribute__((ext_vector_type(8)))  float    v8f;
typedef __attribute__((ext_vector_type(4)))  float    v4f;
typedef __attribute__((ext_vector_type(4)))  unsigned int u32x4;
typedef __attribute__((ext_vector_type(4)))  int      i32x4;
typedef __attribute__((ext_vector_type(8)))  int      i32x8;

#define B_   4
#define S_   2048
#define NU   1024
#define DK   64
#define ROWS (B_ * S_)

union HV16 { v16h v; v8h h[2]; };

__device__ __forceinline__ v8f wmma_f16(v16h a, v16h b, v8f c) {
  // D = A(16x32 f16) x B(32x16 f16) + C(16x16 f32)
  return __builtin_amdgcn_wmma_f32_16x16x32_f16(
      /*neg_a=*/false, a, /*neg_b=*/false, b,
      /*c_mod=*/(short)0, c, /*reuse_a=*/false, /*reuse_b=*/false);
}

// ---------------------------------------------------------------------------
// TDM: issue a 2-D tile load Global->LDS via a hand-built D# descriptor.
//   tensor: rows of `tensor_w` f16 elements, row stride `stride_w` elements;
//   tile:   tile_h rows x tile_w elements starting at `gaddr`, packed
//           contiguously into LDS at byte offset `lds_off`.
// D# layout per CDNA5 ISA (data_size=1 -> 2B).
// ---------------------------------------------------------------------------
__device__ __forceinline__ void tdm_load_2d(
    unsigned lds_off, const void* gaddr,
    unsigned tensor_w, unsigned tensor_h, unsigned stride_w,
    unsigned tile_w, unsigned tile_h)
{
  unsigned long long ga = (unsigned long long)(uintptr_t)gaddr;
  u32x4 g0;
  g0[0] = 1u;                                            // count=1 (valid, user)
  g0[1] = lds_off;                                       // lds_addr
  g0[2] = (unsigned)(ga & 0xFFFFFFFFu);                  // global_addr[31:0]
  g0[3] = (unsigned)((ga >> 32) & 0x01FFFFFFu)           // global_addr[56:32]
        | 0x80000000u;                                   // type=2 ("image")
  i32x8 g1;
  g1[0] = (int)(1u << 16);                               // data_size=1 (2 bytes)
  g1[1] = (int)((tensor_w & 0xFFFFu) << 16);             // tensor_dim0[15:0]
  g1[2] = (int)(((tensor_w >> 16) & 0xFFFFu)
        | ((tensor_h & 0xFFFFu) << 16));                 // dim0 hi | dim1 lo
  g1[3] = (int)(((tensor_h >> 16) & 0xFFFFu)
        | ((tile_w & 0xFFFFu) << 16));                   // dim1 hi | tile_dim0
  g1[4] = (int)(tile_h & 0xFFFFu);                       // tile_dim1 | tile_dim2=0
  g1[5] = (int)stride_w;                                 // tensor_dim0_stride lo
  g1[6] = 0;                                             // stride0 hi | stride1 lo
  g1[7] = 0;
  i32x4 z4 = {0, 0, 0, 0};
#if __has_include(<hip/amd_detail/amd_gfx1250_TDM.h>)
  i32x8 z8 = {0, 0, 0, 0, 0, 0, 0, 0};
  __builtin_amdgcn_tensor_load_to_lds(g0, g1, z4, z4, z8, 0);   // 6-arg toolchain
#else
  __builtin_amdgcn_tensor_load_to_lds(g0, g1, z4, z4, 0);       // 5-arg ROCm 7.2
#endif
}

// ---------------------------------------------------------------------------
// Projection: out[row][d] = X[row][:] . W[d][:] + bias[d]   (then *scale)
// One wave handles a 16-row M tile across all 64 output dims (4 N tiles).
// transposed==0 : store f16 row-major [row][DK]
// transposed==1 : store f16 as Vt[(b*DK+d)][S]  (for P.V B-fragments)
// ---------------------------------------------------------------------------
__global__ __launch_bounds__(256) void proj_kernel(
    const float* __restrict__ X, const float* __restrict__ W,
    const float* __restrict__ bias, _Float16* __restrict__ out,
    float scale, int transposed)
{
  const int wid  = threadIdx.x >> 5;
  const int lane = threadIdx.x & 31;
  const int lr   = lane & 15;
  const int hid  = lane >> 4;           // half-wave id (K split for A/B frags)
  const int tile = blockIdx.x * 8 + wid;   // 0..511
  const int row0 = tile * 16;

  v8f acc[4] = {};                       // 4 N-tiles of 16 dims

  const float* xrow = X + (size_t)(row0 + lr) * NU;

  for (int k0 = 0; k0 < NU; k0 += 32) {
    // --- A fragment: 16x32 slice of X, f32 -> f16 ---
    // lane(hid=0): K = {k0+0..15}; lane(hid=1): {k0+16..31}
    HV16 A;
    {
      const int o0 = k0 + hid * 16;
      const int o1 = k0 + hid * 16 + 8;
      v4f f0 = *(const v4f*)(xrow + o0);
      v4f f1 = *(const v4f*)(xrow + o0 + 4);
      v4f f2 = *(const v4f*)(xrow + o1);
      v4f f3 = *(const v4f*)(xrow + o1 + 4);
#pragma unroll
      for (int i = 0; i < 4; ++i) {
        A.v[i]      = (_Float16)f0[i];
        A.v[4 + i]  = (_Float16)f1[i];
        A.v[8 + i]  = (_Float16)f2[i];
        A.v[12 + i] = (_Float16)f3[i];
      }
    }
#pragma unroll
    for (int nt = 0; nt < 4; ++nt) {
      // --- B fragment: W^T[k0..k0+31][nt*16..nt*16+15] ---
      const float* wrow = W + (size_t)(nt * 16 + lr) * NU + k0 + hid * 16;
      HV16 Bv;
      v4f g0 = *(const v4f*)(wrow);
      v4f g1 = *(const v4f*)(wrow + 4);
      v4f g2 = *(const v4f*)(wrow + 8);
      v4f g3 = *(const v4f*)(wrow + 12);
#pragma unroll
      for (int i = 0; i < 4; ++i) {
        Bv.v[i]      = (_Float16)g0[i];
        Bv.v[4 + i]  = (_Float16)g1[i];
        Bv.v[8 + i]  = (_Float16)g2[i];
        Bv.v[12 + i] = (_Float16)g3[i];
      }
      acc[nt] = wmma_f16(A.v, Bv.v, acc[nt]);
    }
  }

  // Epilogue: D-layout => lane = column, VGPR v = row (v or v+8 per half)
#pragma unroll
  for (int nt = 0; nt < 4; ++nt) {
    const int d  = nt * 16 + lr;
    const float bv = bias[d];
#pragma unroll
    for (int v = 0; v < 8; ++v) {
      const int row = row0 + v + 8 * hid;
      const float val = (acc[nt][v] + bv) * scale;
      if (!transposed) {
        out[(size_t)row * DK + d] = (_Float16)val;
      } else {
        const int b = row / S_, s = row % S_;
        out[((size_t)(b * DK + d)) * S_ + s] = (_Float16)val;
      }
    }
  }
}

// ---------------------------------------------------------------------------
// Attention: one wave per 16-query tile; 32 keys per iteration.
// K/V tiles double-buffered in LDS via TDM (issued one iteration ahead);
// scores = Qh.Kh^T (scale pre-folded into Qh), mask-select, online softmax,
// P staged via LDS (D-layout -> A-layout transpose), out += P.V.
// ---------------------------------------------------------------------------
__global__ __launch_bounds__(256) void attn_kernel(
    const _Float16* __restrict__ Qh, const _Float16* __restrict__ Kh,
    const _Float16* __restrict__ Vt, const int* __restrict__ mask,
    float* __restrict__ out)
{
  __shared__ __align__(128) _Float16 Klds[2][32 * DK];   // 2 x 4 KB
  __shared__ __align__(128) _Float16 Vlds[2][DK * 32];   // 2 x 4 KB
  __shared__ __align__(64)  _Float16 Plds[8][16][32];    // 8 KB per-wave P tiles

  const int wid  = threadIdx.x >> 5;
  const int lane = threadIdx.x & 31;
  const int lr   = lane & 15;
  const int hid  = lane >> 4;
  const int tile = blockIdx.x * 8 + wid;   // 0..511 (block = 8 tiles, 1 batch)
  const int b    = tile >> 7;              // same for all waves in block
  const int q0   = (tile & 127) * 16;      // query row base within batch
  const int grow0 = b * S_ + q0;           // global row base

  const unsigned koff0 = (unsigned)(uintptr_t)(void*)&Klds[0][0];
  const unsigned koff1 = (unsigned)(uintptr_t)(void*)&Klds[1][0];
  const unsigned voff0 = (unsigned)(uintptr_t)(void*)&Vlds[0][0];
  const unsigned voff1 = (unsigned)(uintptr_t)(void*)&Vlds[1][0];

  // Q A-fragments for both D_K chunks (d=0..31, 32..63), loaded once.
  HV16 QA[2];
  {
    const _Float16* qr = Qh + (size_t)(grow0 + lr) * DK;
#pragma unroll
    for (int c = 0; c < 2; ++c) {
      const int k0 = c * 32;
      QA[c].h[0] = *(const v8h*)(qr + k0 + hid * 16);
      QA[c].h[1] = *(const v8h*)(qr + k0 + hid * 16 + 8);
    }
  }

  // prologue: kick off tile 0 into buffer 0
  if (threadIdx.x < 32) {
    tdm_load_2d(koff0, Kh + (size_t)(b * S_) * DK,
                DK, 32, DK, DK, 32);
    tdm_load_2d(voff0, Vt + (size_t)(b * DK) * S_,
                32, DK, S_, 32, DK);
  }

  v8f acc[4] = {};
  float Mr[8], Lr[8];
#pragma unroll
  for (int v = 0; v < 8; ++v) { Mr[v] = -3.0e38f; Lr[v] = 0.0f; }

  for (int kb = 0; kb < S_; kb += 32) {
    const int buf = (kb >> 5) & 1;

    // ---- ping-pong staging: issue tile i+1, retire tile i ----
    __syncthreads();                         // nobody still reads buf^1
    if (threadIdx.x < 32) {
      if (kb + 32 < S_) {
        const unsigned kn = buf ? koff0 : koff1;
        const unsigned vn = buf ? voff0 : voff1;
        tdm_load_2d(kn, Kh + (size_t)(b * S_ + kb + 32) * DK,
                    DK, 32, DK, DK, 32);
        tdm_load_2d(vn, Vt + (size_t)(b * DK) * S_ + kb + 32,
                    32, DK, S_, 32, DK);
        // TDM ops of one wave complete in order: <=2 outstanding means the
        // older pair (tile i) has landed; tile i+1 stays in flight.
        __builtin_amdgcn_s_wait_tensorcnt(2);
      } else {
        __builtin_amdgcn_s_wait_tensorcnt(0);
      }
    }
    __syncthreads();                         // buf published to all waves

    // prefetch next iteration's mask tile (largest HBM stream)
    if (kb + 32 < S_)
      __builtin_prefetch(&mask[((size_t)b * S_ + q0 + lr) * S_ + kb + 32 + hid * 16], 0, 0);

    const _Float16* Kt = &Klds[buf][0];
    const _Float16* Vtl = &Vlds[buf][0];

    // ---- scores: two 16-key tiles via 4 WMMAs (B-frags from LDS) ----
    v8f s0 = {}, s1 = {};
#pragma unroll
    for (int c = 0; c < 2; ++c) {
      const int k0 = c * 32;
      const _Float16* kp0 = Kt + (lr)      * DK + k0 + hid * 16;
      const _Float16* kp1 = Kt + (16 + lr) * DK + k0 + hid * 16;
      HV16 KB0, KB1;
      KB0.h[0] = *(const v8h*)(kp0);  KB0.h[1] = *(const v8h*)(kp0 + 8);
      KB1.h[0] = *(const v8h*)(kp1);  KB1.h[1] = *(const v8h*)(kp1 + 8);
      s0 = wmma_f16(QA[c].v, KB0.v, s0);
      s1 = wmma_f16(QA[c].v, KB1.v, s1);
    }

    // ---- mask + online softmax (row reductions via shfl within 16-lane half)
    float p0[8], p1[8], corr[8];
#pragma unroll
    for (int v = 0; v < 8; ++v) {
      const int row = q0 + v + 8 * hid;
      const size_t mbase = ((size_t)b * S_ + row) * S_ + kb;
      const int m0 = mask[mbase + lr];
      const int m1 = mask[mbase + 16 + lr];
      const float a  = m0 ? s0[v] : -1.0e9f;   // s*m - 1e9*(1-m) for m in {0,1}
      const float c2 = m1 ? s1[v] : -1.0e9f;

      float t = fmaxf(a, c2);
      t = fmaxf(t, __shfl_xor(t, 1, 32));
      t = fmaxf(t, __shfl_xor(t, 2, 32));
      t = fmaxf(t, __shfl_xor(t, 4, 32));
      t = fmaxf(t, __shfl_xor(t, 8, 32));

      const float newM = fmaxf(Mr[v], t);
      const float cf = __expf(Mr[v] - newM);
      const float e0 = __expf(a - newM);
      const float e1 = __expf(c2 - newM);
      float su = e0 + e1;
      su += __shfl_xor(su, 1, 32);
      su += __shfl_xor(su, 2, 32);
      su += __shfl_xor(su, 4, 32);
      su += __shfl_xor(su, 8, 32);

      Lr[v] = Lr[v] * cf + su;
      Mr[v] = newM;
      corr[v] = cf;
      p0[v] = e0;  p1[v] = e1;
    }

    // rescale running accumulator
#pragma unroll
    for (int nt = 0; nt < 4; ++nt)
#pragma unroll
      for (int v = 0; v < 8; ++v)
        acc[nt][v] *= corr[v];

    // ---- D-layout -> A-layout transpose of P through per-wave LDS ----
#pragma unroll
    for (int v = 0; v < 8; ++v) {
      Plds[wid][v + 8 * hid][lr]      = (_Float16)p0[v];
      Plds[wid][v + 8 * hid][16 + lr] = (_Float16)p1[v];
    }
    asm volatile("s_wait_dscnt 0" ::: "memory");   // all lanes' DS stores visible
    HV16 PA;
    {
      const _Float16* pr = &Plds[wid][lr][0];
      PA.h[0] = *(const v8h*)(pr + hid * 16);
      PA.h[1] = *(const v8h*)(pr + hid * 16 + 8);
    }

    // ---- out += P(16x32) . V(32x64) via 4 WMMAs (B-frags from LDS) ----
#pragma unroll
    for (int nt = 0; nt < 4; ++nt) {
      const _Float16* vp = Vtl + (nt * 16 + lr) * 32 + hid * 16;
      HV16 VB;
      VB.h[0] = *(const v8h*)(vp);
      VB.h[1] = *(const v8h*)(vp + 8);
      acc[nt] = wmma_f16(PA.v, VB.v, acc[nt]);
    }
  }

  // ---- epilogue: divide by softmax sum, store f32 ----
#pragma unroll
  for (int v = 0; v < 8; ++v) {
    const float inv = 1.0f / Lr[v];
    const int row = grow0 + v + 8 * hid;
#pragma unroll
    for (int nt = 0; nt < 4; ++nt)
      out[(size_t)row * DK + nt * 16 + lr] = acc[nt][v] * inv;
  }
}

// ---------------------------------------------------------------------------
extern "C" void kernel_launch(void* const* d_in, const int* in_sizes, int n_in,
                              void* d_out, int out_size, void* d_ws, size_t ws_size,
                              hipStream_t stream) {
  (void)in_sizes; (void)n_in; (void)out_size; (void)ws_size;

  const float* Q    = (const float*)d_in[0];
  const float* K    = (const float*)d_in[1];
  const float* V    = (const float*)d_in[2];
  const int*   mask = (const int*)  d_in[3];
  const float* Wq_w = (const float*)d_in[4];
  const float* Wq_b = (const float*)d_in[5];
  const float* Wk_w = (const float*)d_in[6];
  const float* Wk_b = (const float*)d_in[7];
  const float* Wv_w = (const float*)d_in[8];
  const float* Wv_b = (const float*)d_in[9];

  _Float16* Qh = (_Float16*)d_ws;                 // [ROWS][DK]
  _Float16* Kh = Qh + (size_t)ROWS * DK;          // [ROWS][DK]
  _Float16* Vt = Kh + (size_t)ROWS * DK;          // [B*DK][S]
  float* out = (float*)d_out;

  dim3 blk(256);                 // 8 waves (wave32)
  dim3 grid(ROWS / 16 / 8);      // 64 blocks: one 16-row tile per wave

  // scale 1/sqrt(64) folded into Q projection
  proj_kernel<<<grid, blk, 0, stream>>>(Q, Wq_w, Wq_b, Qh, 0.125f, 0);
  proj_kernel<<<grid, blk, 0, stream>>>(K, Wk_w, Wk_b, Kh, 1.0f,   0);
  proj_kernel<<<grid, blk, 0, stream>>>(V, Wv_w, Wv_b, Vt, 1.0f,   1);
  attn_kernel<<<grid, blk, 0, stream>>>(Qh, Kh, Vt, mask, out);
}
